// DiscreteGCNLayer_5540507811925
// MI455X (gfx1250) — matrix operand order repaired and checked by reading, not compiled, not run on hardware
//
#include <hip/hip_runtime.h>

// MI455X roofline: adj(134MB fp32) + x + out ~= 170MB -> ~7.3us at 23.3 TB/s.
// FP32 WMMA (16x16x4) would be compute bound (~30us); f16 WMMA (16x16x32, f32
// accum) runs ~4us of math -> memory bound. So: convert operands to f16 on the
// fly, keep accumulation f32, stream adj with non-temporal loads.

typedef __attribute__((ext_vector_type(4)))  float     v4f;
typedef __attribute__((ext_vector_type(8)))  float     v8f;
typedef __attribute__((ext_vector_type(8)))  _Float16  v8h;
typedef __attribute__((ext_vector_type(16))) _Float16  v16h;

#define BM 128
#define BN 128
#define BK 32
#define LDSS 40          // 32 halves + 8 pad: 80B row stride, conflict-free b128 reads
#define NTHREADS 256

static constexpr int MDIM = 2048;   // rows per batch (N in the problem)
static constexpr int NDIM = 256;    // output feature dim
static constexpr int BATCH = 8;

union FragH { v16h v; v8h h[2]; };
union AccU  { v8f  v; float f[8]; };

// ws layout: wT[256][256] ternary f16 (transposed weight), then sT[b][o][m] f16
__global__ void quantize_weight(const float* __restrict__ W, _Float16* __restrict__ wT) {
  const int o = blockIdx.x;     // 0..255
  const int k = threadIdx.x;    // 0..255
  const float w = W[k * NDIM + o];
  const float q = (w > 0.01f ? 1.0f : 0.0f) - (w < -0.01f ? 1.0f : 0.0f);
  wT[o * NDIM + k] = (_Float16)q;   // exact +/-1 or 0; 0.01 scale applied later in f32
}

// MODE 0: support^T = (x @ wT^T) * scale, f16 output transposed [o][m]
// MODE 1: out = relu(adj @ sT^T + bias), f32 output [m][o]
template<int MODE, int KDIM>
__global__ __launch_bounds__(NTHREADS)
void gcn_gemm(const float* __restrict__ A, const _Float16* __restrict__ B,
              const float* __restrict__ bias, void* __restrict__ Out, float scale)
{
  const int tid = threadIdx.x;
  const int bM  = blockIdx.x * BM;
  const int bN  = blockIdx.y * BN;
  const int b   = blockIdx.z;

  const float*    Ab = A + (size_t)b * MDIM * KDIM;
  const _Float16* Bb = B + (MODE == 1 ? (size_t)b * NDIM * KDIM : (size_t)0);

  __shared__ alignas(16) _Float16 ldsA[BM * LDSS];
  __shared__ alignas(16) _Float16 ldsB[BN * LDSS];

  // tile-load mapping: 2 threads per row, each moves 16 contiguous K elements
  const int lrow = tid >> 1;
  const int lseg = (tid & 1) * 16;

  const int lane = tid & 31;
  const int wm = ((tid >> 5) >> 1) * 32;   // wave M offset: 0,32,64,96
  const int wn = ((tid >> 5) & 1) * 64;    // wave N offset: 0,64

  AccU acc[2][4];
  #pragma unroll
  for (int mi = 0; mi < 2; ++mi)
    #pragma unroll
    for (int ni = 0; ni < 4; ++ni)
      #pragma unroll
      for (int j = 0; j < 8; ++j) acc[mi][ni].f[j] = 0.0f;

  v4f areg[4];   // 16 floats of adj/x (fp32)
  v8h breg[2];   // 16 halves of B

  auto globalLoad = [&](int kb) {
    const float* pa = Ab + (size_t)(bM + lrow) * KDIM + kb + lseg;
    #pragma unroll
    for (int i = 0; i < 4; ++i) {
      if (MODE == 1) areg[i] = __builtin_nontemporal_load((const v4f*)(pa + 4 * i));
      else           areg[i] = *(const v4f*)(pa + 4 * i);
    }
    const _Float16* pb = Bb + (size_t)(bN + lrow) * KDIM + kb + lseg;
    breg[0] = *(const v8h*)pb;
    breg[1] = *(const v8h*)(pb + 8);
  };

  auto ldsStore = [&]() {
    _Float16* da = ldsA + lrow * LDSS + lseg;
    v8h h0, h1;
    #pragma unroll
    for (int j = 0; j < 4; ++j) {
      h0[j]     = (_Float16)areg[0][j];
      h0[j + 4] = (_Float16)areg[1][j];
      h1[j]     = (_Float16)areg[2][j];
      h1[j + 4] = (_Float16)areg[3][j];
    }
    *(v8h*)da       = h0;
    *(v8h*)(da + 8) = h1;
    _Float16* db = ldsB + lrow * LDSS + lseg;
    *(v8h*)db       = breg[0];
    *(v8h*)(db + 8) = breg[1];
  };

  constexpr int NT = KDIM / BK;
  globalLoad(0);
  ldsStore();
  __syncthreads();

  for (int kt = 0; kt < NT; ++kt) {
    if (kt + 1 < NT) globalLoad((kt + 1) * BK);   // prefetch next tile into regs

    // A fragment (16x32 f16): lane-half -> K 0-7 / 8-15 in h[0], +16 in h[1]
    FragH fa[2], fb[4];
    #pragma unroll
    for (int mi = 0; mi < 2; ++mi) {
      const _Float16* p = ldsA + (wm + mi * 16 + (lane & 15)) * LDSS + (lane >> 4) * 8;
      fa[mi].h[0] = *(const v8h*)p;
      fa[mi].h[1] = *(const v8h*)(p + 16);
    }
    // B fragment (32x16 f16): lane-half -> K 0-7 / 8-15 in h[0], +16 in h[1]
    #pragma unroll
    for (int ni = 0; ni < 4; ++ni) {
      const _Float16* p = ldsB + (wn + ni * 16 + (lane & 15)) * LDSS + (lane >> 4) * 8;
      fb[ni].h[0] = *(const v8h*)p;
      fb[ni].h[1] = *(const v8h*)(p + 16);
    }
    #pragma unroll
    for (int mi = 0; mi < 2; ++mi)
      #pragma unroll
      for (int ni = 0; ni < 4; ++ni)
        acc[mi][ni].v = __builtin_amdgcn_wmma_f32_16x16x32_f16(
            false, fa[mi].v, false, fb[ni].v, (short)0, acc[mi][ni].v, false, false);

    __syncthreads();
    if (kt + 1 < NT) {
      ldsStore();
      __syncthreads();
    }
  }

  // D fragment: lanes 0-15 -> N=lane, M=vgpr; lanes 16-31 -> N=lane-16, M=vgpr+8
  if constexpr (MODE == 0) {
    _Float16* OutT = (_Float16*)Out + (size_t)b * NDIM * MDIM;   // sT[b][o][m]
    #pragma unroll
    for (int mi = 0; mi < 2; ++mi)
      #pragma unroll
      for (int ni = 0; ni < 4; ++ni) {
        const int o  = bN + wn + ni * 16 + (lane & 15);
        const int m0 = bM + wm + mi * 16 + (lane >> 4) * 8;
        v8h h;
        #pragma unroll
        for (int r = 0; r < 8; ++r) h[r] = (_Float16)(acc[mi][ni].f[r] * scale);
        *(v8h*)(OutT + (size_t)o * MDIM + m0) = h;   // 8 contiguous m's: one b128
      }
  } else {
    float* O = (float*)Out + (size_t)b * MDIM * NDIM;
    #pragma unroll
    for (int ni = 0; ni < 4; ++ni) {
      const int o  = bN + wn + ni * 16 + (lane & 15);
      const float bv = bias[o];
      #pragma unroll
      for (int mi = 0; mi < 2; ++mi) {
        const int m0 = bM + wm + mi * 16 + (lane >> 4) * 8;
        #pragma unroll
        for (int r = 0; r < 8; ++r) {
          float v = acc[mi][ni].f[r] + bv;
          v = v > 0.0f ? v : 0.0f;
          __builtin_nontemporal_store(v, O + (size_t)(m0 + r) * NDIM + o);
        }
      }
    }
  }
}

extern "C" void kernel_launch(void* const* d_in, const int* in_sizes, int n_in,
                              void* d_out, int out_size, void* d_ws, size_t ws_size,
                              hipStream_t stream) {
  const float* x      = (const float*)d_in[0];   // [8, 2048, 256]
  const float* adj    = (const float*)d_in[1];   // [8, 2048, 2048]
  const float* weight = (const float*)d_in[2];   // [256, 256]
  const float* bias   = (const float*)d_in[3];   // [256]
  float* out = (float*)d_out;                    // [8, 2048, 256]

  _Float16* wT = (_Float16*)d_ws;                // 256*256 f16 = 128KB
  _Float16* sT = wT + NDIM * NDIM;               // 8*256*2048 f16 = 8MB

  quantize_weight<<<dim3(NDIM), dim3(NDIM), 0, stream>>>(weight, wT);

  dim3 grid(MDIM / BM, NDIM / BN, BATCH);        // (16, 2, 8)
  gcn_gemm<0, 256 ><<<grid, NTHREADS, 0, stream>>>(x,   wT, nullptr, sT,  0.01f);
  gcn_gemm<1, 2048><<<grid, NTHREADS, 0, stream>>>(adj, sT, bias,    out, 1.0f);
}
